// Attention_13048110645268
// MI455X (gfx1250) — matrix operand order repaired and checked by reading, not compiled, not run on hardware
//
#include <hip/hip_runtime.h>
#include <math.h>

// ---------------------------------------------------------------------------
// Types
// ---------------------------------------------------------------------------
typedef __bf16 bf16;
typedef __attribute__((ext_vector_type(16))) __bf16 v16bf;
typedef __attribute__((ext_vector_type(8)))  __bf16 v8bf;
typedef __attribute__((ext_vector_type(8)))  float  v8f;
typedef __attribute__((ext_vector_type(4)))  unsigned int v4u;
typedef __attribute__((ext_vector_type(8)))  int v8i;
typedef __attribute__((ext_vector_type(4)))  int v4i;

union U16BF { v16bf v; v8bf h[2]; };

// D = A(16x32 bf16) * B(32x16 bf16) + C(16x16 f32)
__device__ __forceinline__ v8f wmma_bf16(v16bf a, v16bf b, v8f c) {
    return __builtin_amdgcn_wmma_f32_16x16x32_bf16(
        false, a, false, b, (short)0, c, false, false);
}

// A operand fragment (16x32) from row-major [row][k] storage, leading dim ld.
__device__ __forceinline__ v16bf load_frag_a(const bf16* p, int ld, int lane) {
    int r  = lane & 15;
    int hf = lane >> 4;
    const bf16* q = p + r * ld + 8 * hf;
    U16BF u;
    u.h[0] = *(const v8bf*)(q);
    u.h[1] = *(const v8bf*)(q + 16);
    return u.v;
}

// B operand fragment (32x16, K x N) from TRANSPOSED storage Bt[n][k].
__device__ __forceinline__ v16bf load_frag_b(const bf16* p, int ld, int lane) {
    int n  = lane & 15;
    int hf = lane >> 4;
    const bf16* q = p + n * ld + 8 * hf;
    U16BF u;
    u.h[0] = *(const v8bf*)(q);
    u.h[1] = *(const v8bf*)(q + 16);
    return u.v;
}

// ---------------------------------------------------------------------------
// DPP ROW_XMASK butterfly (lane <- lane ^ n within each row of 16):
// pure-VALU 16-lane reductions (no LDS / ds_bpermute round trips).
// ---------------------------------------------------------------------------
template <int CTRL>
__device__ __forceinline__ float dpp_bfly(float v) {
    union { float f; int i; } u, r;
    u.f = v;
    r.i = __builtin_amdgcn_update_dpp(u.i, u.i, CTRL, 0xf, 0xf, true);
    return r.f;
}
__device__ __forceinline__ float red_max16(float v) {
    v = fmaxf(v, dpp_bfly<0x161>(v));   // xor 1
    v = fmaxf(v, dpp_bfly<0x162>(v));   // xor 2
    v = fmaxf(v, dpp_bfly<0x164>(v));   // xor 4
    v = fmaxf(v, dpp_bfly<0x168>(v));   // xor 8
    return v;
}
__device__ __forceinline__ float red_sum16(float v) {
    v += dpp_bfly<0x161>(v);
    v += dpp_bfly<0x162>(v);
    v += dpp_bfly<0x164>(v);
    v += dpp_bfly<0x168>(v);
    return v;
}

// ---------------------------------------------------------------------------
// Tensor Data Mover: DMA a 2-D bf16 tile (tile1 rows x tile0 elems) from
// global [.. , stride0] into LDS at lds_addr, with HW row padding.
//   pad_k:   pad_interval code, pads after 2^(pad_k+1) dwords
//   pad_amt: pad_amount code, inserts (pad_amt+1) dwords of padding
// D# layout per CDNA5 ISA 8.3/8.4; groups 2/3 zero (2-D tile, no iterate).
// ---------------------------------------------------------------------------
__device__ __forceinline__ void tdm_load_2d(unsigned int lds_addr,
                                            const bf16* gp,
                                            unsigned int tensor_d0,
                                            unsigned int tensor_d1,
                                            unsigned int tile0,
                                            unsigned int tile1,
                                            unsigned int stride0,
                                            unsigned int pad_k,
                                            unsigned int pad_amt) {
    unsigned long long ga = (unsigned long long)gp;
    v4u g0;
    g0.x = 1u;                                   // count=1, user descriptor
    g0.y = lds_addr;                             // LDS byte address
    g0.z = (unsigned int)(ga & 0xFFFFFFFFu);     // global_addr[31:0]
    g0.w = (unsigned int)((ga >> 32) & 0x1FFFFFFu) | (2u << 30);  // [56:32]|type=2
    v8i g1;
    g1[0] = (int)((1u << 16)            // data_size = 2 bytes
                | (1u << 20)            // pad_enable
                | (pad_k << 22)
                | (pad_amt << 25));
    g1[1] = (int)((tensor_d0 & 0xFFFFu) << 16);                    // td0 lo16
    g1[2] = (int)(((tensor_d0 >> 16) & 0xFFFFu) | ((tensor_d1 & 0xFFFFu) << 16));
    g1[3] = (int)(((tensor_d1 >> 16) & 0xFFFFu) | ((tile0 & 0xFFFFu) << 16));
    g1[4] = (int)(tile1 & 0xFFFFu);                                // tile_dim1 (dim2=0)
    g1[5] = (int)stride0;                                          // stride0 lo32
    g1[6] = 0;                                                     // stride0 hi / stride1 lo
    g1[7] = 0;
    v4i g2 = {0, 0, 0, 0};
    v4i g3 = {0, 0, 0, 0};
    v8i g4 = {0, 0, 0, 0, 0, 0, 0, 0};
    __builtin_amdgcn_tensor_load_to_lds(g0, g1, g2, g3, g4, 0);
}

__device__ __forceinline__ unsigned int lds_addr_of(const void* p) {
    // generic "shared" address: high 32 = shared aperture, low 32 = LDS offset
    return (unsigned int)(unsigned long long)p;
}

// ---------------------------------------------------------------------------
// Conversion pre-passes
// ---------------------------------------------------------------------------
__global__ void cvt_f32_bf16(const float* __restrict__ in, bf16* __restrict__ out,
                             int n) {
    int i = blockIdx.x * 256 + threadIdx.x;
    if (i < n) out[i] = (bf16)in[i];
}

// out[n*K + k] = (bf16) in[k*N + n]   (W [K][N] -> Wt [N][K])
__global__ void transpose_cvt(const float* __restrict__ in, bf16* __restrict__ out,
                              int K, int N) {
    long i = (long)blockIdx.x * 256 + threadIdx.x;
    long total = (long)K * N;
    if (i >= total) return;
    int k = (int)(i / N);
    int n = (int)(i % N);
    out[(long)n * K + k] = (bf16)in[i];
}

// ---------------------------------------------------------------------------
// GEMM:  C[M,N] = A[M,K](bf16) * Wt[N,K](bf16)^T + bias
//   MODE 0: write f32 to outF[M][N]
//   MODE 1: scatter bf16 -> Q (pre-scaled by 1/sqrt(64)), K [B,H,S,d];
//           V stored TRANSPOSED [B,H,d,S]
// Block tile 128(M) x 64(N), 256 threads = 8 waves, wave tile 32x32.
// K-step 64: A/W tiles staged by TDM with HW padding (row 128B + 16B pad ->
// stride 72), double-buffered; 8 WMMAs per tensorcnt-wait + barrier.
// ---------------------------------------------------------------------------
template <int MODE>
__global__ __launch_bounds__(256) void gemm_bf16(
    const bf16* __restrict__ A, const bf16* __restrict__ Wt,
    const float* __restrict__ bias, float* __restrict__ outF,
    bf16* __restrict__ Qo, bf16* __restrict__ Ko, bf16* __restrict__ Vo,
    int K, int N) {
    __shared__ __align__(16) bf16 Al[2][128 * 72];
    __shared__ __align__(16) bf16 Wl[2][64 * 72];

    const int tid  = threadIdx.x;
    const int lane = tid & 31;
    const int wid  = tid >> 5;
    const int wm   = wid & 3;
    const int wn   = wid >> 2;
    const int n16  = lane & 15;
    const int half = lane >> 4;
    const bool w0  = (wid == 0);

    const int rowBlk = blockIdx.y * 128;
    const int colBlk = blockIdx.x * 64;

    const bf16* Ablk = A + (long)rowBlk * K;
    const bf16* Wblk = Wt + (long)colBlk * K;
    unsigned int alds[2] = {lds_addr_of(&Al[0][0]), lds_addr_of(&Al[1][0])};
    unsigned int wlds[2] = {lds_addr_of(&Wl[0][0]), lds_addr_of(&Wl[1][0])};

    const v8f vzero = {0.f, 0.f, 0.f, 0.f, 0.f, 0.f, 0.f, 0.f};
    v8f acc[2][2] = {{vzero, vzero}, {vzero, vzero}};

    if (w0) {  // prologue DMA: tiles for kk=0 into buffer 0
        tdm_load_2d(alds[0], Ablk, K, 4096, 64, 128, K, /*pad 32dw+4dw*/ 4, 3);
        tdm_load_2d(wlds[0], Wblk, K, N,    64, 64,  K, 4, 3);
    }

    int ib = 0;
    for (int kk = 0; kk < K; kk += 64, ib ^= 1) {
        if (w0) __builtin_amdgcn_s_wait_tensorcnt(0);
        __syncthreads();  // current tiles visible; previous buffer reads retired
        if (w0 && kk + 64 < K) {  // prefetch next tiles, overlapping compute
            tdm_load_2d(alds[ib ^ 1], Ablk + kk + 64, K, 4096, 64, 128, K, 4, 3);
            tdm_load_2d(wlds[ib ^ 1], Wblk + kk + 64, K, N,    64, 64,  K, 4, 3);
        }

        v16bf am[2][2], bn[2][2];
#pragma unroll
        for (int mt = 0; mt < 2; ++mt)
#pragma unroll
            for (int c = 0; c < 2; ++c)
                am[mt][c] = load_frag_a(&Al[ib][(wm * 32 + mt * 16) * 72 + 32 * c], 72, lane);
#pragma unroll
        for (int nt = 0; nt < 2; ++nt)
#pragma unroll
            for (int c = 0; c < 2; ++c)
                bn[nt][c] = load_frag_b(&Wl[ib][(wn * 32 + nt * 16) * 72 + 32 * c], 72, lane);
#pragma unroll
        for (int mt = 0; mt < 2; ++mt)
#pragma unroll
            for (int nt = 0; nt < 2; ++nt)
#pragma unroll
                for (int c = 0; c < 2; ++c)
                    acc[mt][nt] = wmma_bf16(am[mt][c], bn[nt][c], acc[mt][nt]);
    }

#pragma unroll
    for (int mt = 0; mt < 2; ++mt)
#pragma unroll
        for (int nt = 0; nt < 2; ++nt)
#pragma unroll
            for (int r = 0; r < 8; ++r) {
                int gm = rowBlk + wm * 32 + mt * 16 + r + 8 * half;
                int gn = colBlk + wn * 32 + nt * 16 + n16;
                float v = acc[mt][nt][r] + bias[gn];
                if (MODE == 0) {
                    outF[(long)gm * N + gn] = v;
                } else {
                    int chunk = gn >> 10;        // 0=Q 1=K 2=V
                    int rem   = gn & 1023;
                    int h     = rem >> 6;
                    int dc    = rem & 63;
                    int b     = gm >> 11;
                    int s     = gm & 2047;
                    long bh   = b * 16 + h;
                    if (chunk == 0) {
                        // fold softmax scale 1/sqrt(64) into Q once here
                        Qo[(bh * 2048 + s) * 64 + dc] = (bf16)(v * 0.125f);
                    } else if (chunk == 1) {
                        Ko[(bh * 2048 + s) * 64 + dc] = (bf16)v;
                    } else {
                        Vo[(bh * 64 + dc) * 2048 + s] = (bf16)v;  // V^T
                    }
                }
            }
}

// ---------------------------------------------------------------------------
// Flash attention: softmax((Q/8) K^T) V  per (batch, head); Q pre-scaled.
// Grid: (S/128, B*H). 8 waves; wave owns 16 q-rows x d=64.
// K [S,d] and V^T [d,S] chunk tiles staged by TDM (row 128B + 16B pad ->
// stride 72), double-buffered. Softmax reductions are DPP ROW_XMASK.
// ---------------------------------------------------------------------------
__global__ __launch_bounds__(256) void attn_kernel(
    const bf16* __restrict__ Q, const bf16* __restrict__ K,
    const bf16* __restrict__ Vt, bf16* __restrict__ ctx) {
    __shared__ __align__(16) bf16 Klds[2][64 * 72];   // [kpos][d]
    __shared__ __align__(16) bf16 Vlds[2][64 * 72];   // [d][kpos]
    __shared__ __align__(16) bf16 Plds[8 * 16 * 72];  // per-wave P patch [16][64]

    const int tid  = threadIdx.x;
    const int lane = tid & 31;
    const int wid  = tid >> 5;
    const int n16  = lane & 15;
    const int half = lane >> 4;
    const bool w0  = (wid == 0);

    const int bh = blockIdx.y;  // b*16 + h
    const int b  = bh >> 4;
    const int h  = bh & 15;
    const int q0 = blockIdx.x * 128 + wid * 16;

    const bf16* Qb = Q + ((long)bh * 2048 + q0) * 64;
    const bf16* Kb = K + (long)bh * 2048 * 64;
    const bf16* Vb = Vt + (long)bh * 64 * 2048;
    unsigned int klds[2] = {lds_addr_of(&Klds[0][0]), lds_addr_of(&Klds[1][0])};
    unsigned int vlds[2] = {lds_addr_of(&Vlds[0][0]), lds_addr_of(&Vlds[1][0])};

    // Q fragments held in registers for the whole pass
    v16bf aq[2];
    aq[0] = load_frag_a(Qb, 64, lane);
    aq[1] = load_frag_a(Qb + 32, 64, lane);

    const v8f vzero = {0.f, 0.f, 0.f, 0.f, 0.f, 0.f, 0.f, 0.f};
    v8f o[4] = {vzero, vzero, vzero, vzero};
    float mrun[8], lrun[8];
#pragma unroll
    for (int r = 0; r < 8; ++r) { mrun[r] = -3.0e38f; lrun[r] = 0.f; }

    bf16* pw = Plds + wid * 16 * 72;

    if (w0) {  // prologue DMA for kb=0
        tdm_load_2d(klds[0], Kb, 64, 2048, 64, 64, 64,   /*pad 32dw+4dw*/ 4, 3);
        tdm_load_2d(vlds[0], Vb, 2048, 64, 64, 64, 2048, 4, 3);
    }

    int ib = 0;
    for (int kb = 0; kb < 2048; kb += 64, ib ^= 1) {
        if (w0) __builtin_amdgcn_s_wait_tensorcnt(0);
        __syncthreads();
        if (w0 && kb + 64 < 2048) {  // prefetch next K/V^T chunk
            tdm_load_2d(klds[ib ^ 1], Kb + (long)(kb + 64) * 64, 64, 2048, 64, 64, 64, 4, 3);
            tdm_load_2d(vlds[ib ^ 1], Vb + kb + 64, 2048, 64, 64, 64, 2048, 4, 3);
        }

        // ---- S = Q K^T : 16 x 64 in 4 C-tiles (Q carries the 1/8 scale) ----
        v8f st[4];
#pragma unroll
        for (int nt = 0; nt < 4; ++nt) {
            v8f s = vzero;
#pragma unroll
            for (int c = 0; c < 2; ++c) {
                v16bf bk = load_frag_b(&Klds[ib][(nt * 16) * 72 + 32 * c], 72, lane);
                s = wmma_bf16(aq[c], bk, s);
            }
            st[nt] = s;
        }

        // ---- online softmax (DPP 16-lane reductions) -----------------------
#pragma unroll
        for (int r = 0; r < 8; ++r) {
            float mx = fmaxf(fmaxf(st[0][r], st[1][r]), fmaxf(st[2][r], st[3][r]));
            mx = red_max16(mx);
            float mnew  = fmaxf(mrun[r], mx);
            float alpha = __expf(mrun[r] - mnew);
            float rs = 0.f;
#pragma unroll
            for (int nt = 0; nt < 4; ++nt) {
                float p = __expf(st[nt][r] - mnew);
                st[nt][r] = p;
                rs += p;
            }
            rs = red_sum16(rs);
            lrun[r] = lrun[r] * alpha + rs;
            mrun[r] = mnew;
#pragma unroll
            for (int dt = 0; dt < 4; ++dt) o[dt][r] *= alpha;
        }

        // ---- P (C layout) -> per-wave LDS patch -> A fragments -------------
#pragma unroll
        for (int nt = 0; nt < 4; ++nt)
#pragma unroll
            for (int r = 0; r < 8; ++r)
                pw[(r + 8 * half) * 72 + nt * 16 + n16] = (bf16)st[nt][r];
        // same-wave LDS ops retire in order; pw is private to this wave

        // ---- O += P V ------------------------------------------------------
#pragma unroll
        for (int c = 0; c < 2; ++c) {
            v16bf ap = load_frag_a(&pw[32 * c], 72, lane);
#pragma unroll
            for (int dt = 0; dt < 4; ++dt) {
                v16bf bv = load_frag_b(&Vlds[ib][(dt * 16) * 72 + 32 * c], 72, lane);
                o[dt] = wmma_bf16(ap, bv, o[dt]);
            }
        }
    }

    // ---- epilogue: O / l  -> ctx[B][S][h*64 + d] (bf16) --------------------
#pragma unroll
    for (int r = 0; r < 8; ++r) {
        float inv = 1.0f / lrun[r];
        long row = (long)b * 2048 + q0 + r + 8 * half;
#pragma unroll
        for (int dt = 0; dt < 4; ++dt)
            ctx[row * 1024 + h * 64 + dt * 16 + n16] = (bf16)(o[dt][r] * inv);
    }
}

// ---------------------------------------------------------------------------
// Launch
// ---------------------------------------------------------------------------
extern "C" void kernel_launch(void* const* d_in, const int* in_sizes, int n_in,
                              void* d_out, int out_size, void* d_ws, size_t ws_size,
                              hipStream_t stream) {
    const float* x     = (const float*)d_in[0];  // [2,2048,1024]
    const float* Wqkv  = (const float*)d_in[1];  // [1024,3072]
    const float* bqkv  = (const float*)d_in[2];  // [3072]
    const float* Wproj = (const float*)d_in[3];  // [1024,1024]
    const float* bproj = (const float*)d_in[4];  // [1024]
    float* out = (float*)d_out;                  // [2,2048,1024] f32

    const long M = 4096, D = 1024;
    size_t off = 0;
    auto carve = [&](size_t bytes) {
        char* p = (char*)d_ws + off;
        off += (bytes + 255) & ~(size_t)255;
        return p;
    };
    bf16* xb     = (bf16*)carve(M * D * 2);          // 8 MB
    bf16* wqkvT  = (bf16*)carve(3072l * 1024 * 2);   // 6 MB
    bf16* wprojT = (bf16*)carve(1024l * 1024 * 2);   // 2 MB
    bf16* Qb     = (bf16*)carve(M * 64l * 16 * 2);   // 8 MB (pre-scaled by 1/8)
    bf16* Kb     = (bf16*)carve(M * 64l * 16 * 2);   // 8 MB
    bf16* Vb     = (bf16*)carve(M * 64l * 16 * 2);   // 8 MB (stored transposed)
    bf16* ctx    = (bf16*)carve(M * D * 2);          // 8 MB
    if (off > ws_size) return;

    // 1) convert / transpose to bf16
    cvt_f32_bf16<<<(int)(M * D / 256), 256, 0, stream>>>(x, xb, (int)(M * D));
    transpose_cvt<<<(int)(1024l * 3072 / 256), 256, 0, stream>>>(Wqkv, wqkvT, 1024, 3072);
    transpose_cvt<<<(int)(1024l * 1024 / 256), 256, 0, stream>>>(Wproj, wprojT, 1024, 1024);

    // 2) QKV GEMM -> per-head Q(scaled)/K (row) + V^T (col) bf16
    gemm_bf16<1><<<dim3(3072 / 64, 4096 / 128), 256, 0, stream>>>(
        xb, wqkvT, bqkv, nullptr, Qb, Kb, Vb, 1024, 3072);

    // 3) flash attention -> ctx (bf16, [B,S,H*d])
    attn_kernel<<<dim3(2048 / 128, 32), 256, 0, stream>>>(Qb, Kb, Vb, ctx);

    // 4) output projection -> f32
    gemm_bf16<0><<<dim3(1024 / 64, 4096 / 128), 256, 0, stream>>>(
        ctx, wprojT, bproj, out, nullptr, nullptr, nullptr, 1024, 1024);
}
